// CGpool_6708738916910
// MI455X (gfx1250) — compile-verified, matrix operand order checked by reading, not run on hardware
//
#include <hip/hip_runtime.h>
#include <hip/hip_bf16.h>
#include <math.h>

// ---- problem constants (match reference) ----
#define BB    16
#define NN    1024
#define FF    128
#define NCG   64
#define NCONV 3
#define ROWS  (BB * NN)          // 16384
#define EPSV  0.001f

typedef __attribute__((ext_vector_type(16))) _Float16 v16h;
typedef __attribute__((ext_vector_type(8)))  _Float16 v8h;
typedef __attribute__((ext_vector_type(8)))  float    v8f;

// -------------------------------------------------------------------------
// Weight pre-pack: f32 W[K x Ncols] -> f16 lane-striped B tiles for
// v_wmma_f32_16x16x32_f16.  dst index = ((nt*KT + kt)*32 + lane)*16 + e
// with element e of lane L holding W[k][n], k = kt*32 + (L/16)*8 + (e%8) +
// (e/8)*16, n = nt*16 + (L%16).  One contiguous 32B v16h load per lane/kt.
// -------------------------------------------------------------------------
__global__ void k_pack_w(const float* __restrict__ src, _Float16* __restrict__ dst,
                         int K, int Ncols) {
  int tid = blockIdx.x * blockDim.x + threadIdx.x;
  int total = K * Ncols;
  if (tid >= total) return;
  int e  = tid & 15;
  int L  = (tid >> 4) & 31;
  int KT = K >> 5;
  int kt = (tid >> 9) % KT;
  int nt = tid / (512 * KT);
  int kidx = ((L >> 4) << 3) + (e & 7) + ((e >> 3) << 4);
  int k = kt * 32 + kidx;
  int n = nt * 16 + (L & 15);
  dst[tid] = (_Float16)src[k * Ncols + n];
}

// ---- embedding gather: h = emb[atoms]; keep f32 + f16 copies ----
__global__ void k_embed(const int* __restrict__ atoms, const float* __restrict__ emb,
                        float* __restrict__ h32, _Float16* __restrict__ h16) {
  int tid = blockIdx.x * blockDim.x + threadIdx.x;
  if (tid >= ROWS * FF) return;
  int row = tid >> 7, f = tid & 127;
  float v = emb[atoms[row] * FF + f];
  h32[tid] = v;
  h16[tid] = (_Float16)v;
}

// ---- dense adjacency + degree from bond list ----
__global__ void k_adj(const int* __restrict__ bonds, int nb,
                      float* __restrict__ adj, float* __restrict__ deg) {
  int t = blockIdx.x * blockDim.x + threadIdx.x;
  if (t >= nb) return;
  int b = bonds[3 * t], s = bonds[3 * t + 1], u = bonds[3 * t + 2];
  adj[((size_t)b * NN + s) * NN + u] = 1.0f;
  adj[((size_t)b * NN + u) * NN + s] = 1.0f;
  atomicAdd(deg + b * NN + s, 1.0f);
  atomicAdd(deg + b * NN + u, 1.0f);
}

// -------------------------------------------------------------------------
// WMMA GEMM: out[M x Ncols] = A[M x K](f16) * Bpack(f16) + bias, one wave
// per 16x16 tile, K in steps of 32 via v_wmma_f32_16x16x32_f16.
// MODE 0: out32 = acc + bias.   MODE 1: out16 = tanh(acc + bias) (f16).
// -------------------------------------------------------------------------
template <int MODE>
__global__ __launch_bounds__(32)
void k_gemm_wmma(const _Float16* __restrict__ A, const _Float16* __restrict__ Bp,
                 const float* __restrict__ bias, float* __restrict__ out32,
                 _Float16* __restrict__ out16, int tilesN, int K, int Ncols) {
  const int lane = threadIdx.x;
  const int bid  = blockIdx.x;
  const int nt   = bid % tilesN;
  const int tm   = bid / tilesN;
  const int KT   = K >> 5;
  const int row  = tm * 16 + (lane & 15);
  const int abase = (lane >> 4) * 8;

  v8f acc = {};
  for (int kt = 0; kt < KT; ++kt) {
    const v8h* pa = (const v8h*)(A + (size_t)row * K + kt * 32 + abase);
    v8h a0 = pa[0];     // K = base .. base+7
    v8h a1 = pa[2];     // K = base+16 .. base+23
    v16h av;
#pragma unroll
    for (int i = 0; i < 8; ++i) { av[i] = a0[i]; av[i + 8] = a1[i]; }
    const v16h* pb = (const v16h*)(Bp + ((size_t)(nt * KT + kt) * 32 + lane) * 16);
    v16h bv = pb[0];
    acc = __builtin_amdgcn_wmma_f32_16x16x32_f16(false, av, false, bv,
                                                 (short)0, acc, false, false);
  }
  const int n = nt * 16 + (lane & 15);
  const int mbase = (lane >> 4) * 8;
  const float bn = bias[n];
#pragma unroll
  for (int v = 0; v < 8; ++v) {
    int m = tm * 16 + mbase + v;
    float r = acc[v] + bn;
    if (MODE == 0) out32[(size_t)m * Ncols + n] = r;
    else           out16[(size_t)m * Ncols + n] = (_Float16)tanhf(r);
  }
}

// ---- neighbor aggregation via bond scatter: dh[j] += ch[i] (symmetric) ----
__global__ void k_scatter(const int* __restrict__ bonds, int nb,
                          const float* __restrict__ ch, float* __restrict__ dh) {
  int tid = blockIdx.x * blockDim.x + threadIdx.x;
  if (tid >= nb * FF) return;
  int t = tid >> 7, f = tid & 127;
  int b = bonds[3 * t], s = bonds[3 * t + 1], u = bonds[3 * t + 2];
  size_t rs = ((size_t)b * NN + s) * FF + f;
  size_t ru = ((size_t)b * NN + u) * FF + f;
  atomicAdd(dh + ru, ch[rs]);
  atomicAdd(dh + rs, ch[ru]);
}

// ---- h += dh / deg ; refresh f16 mirror ----
__global__ void k_update(float* __restrict__ h32, _Float16* __restrict__ h16,
                         const float* __restrict__ dh, const float* __restrict__ deg) {
  int tid = blockIdx.x * blockDim.x + threadIdx.x;
  if (tid >= ROWS * FF) return;
  int row = tid >> 7;
  float v = h32[tid] + dh[tid] / deg[row];
  h32[tid] = v;
  h16[tid] = (_Float16)v;
}

// ---- gumbel-softmax rows + column-sum accumulation ----
__global__ void k_softmax(const float* __restrict__ logits, const float* __restrict__ gu,
                          const float* __restrict__ tau_p, float* __restrict__ M,
                          float* __restrict__ colsum) {
  int row = blockIdx.x * blockDim.x + threadIdx.x;
  if (row >= ROWS) return;
  float inv_tau = 1.0f / tau_p[0];
  int b = row >> 10;
  const float* lg = logits + (size_t)row * NCG;
  const float* uu = gu + (size_t)row * NCG;
  float mx = -3.0e38f;
  for (int j = 0; j < NCG; ++j) {
    float z = (lg[j] - logf(-logf(uu[j]))) * inv_tau;
    mx = fmaxf(mx, z);
  }
  float s = 0.0f;
  for (int j = 0; j < NCG; ++j) {
    float z = (lg[j] - logf(-logf(uu[j]))) * inv_tau;
    float e = expf(z - mx);
    M[(size_t)row * NCG + j] = e;
    s += e;
  }
  float inv = 1.0f / s;
  for (int j = 0; j < NCG; ++j) {
    float m = M[(size_t)row * NCG + j] * inv;
    M[(size_t)row * NCG + j] = m;
    atomicAdd(colsum + b * NCG + j, m);
  }
}

__global__ void k_mnorm(const float* __restrict__ M, const float* __restrict__ colsum,
                        float* __restrict__ Mn) {
  int tid = blockIdx.x * blockDim.x + threadIdx.x;
  if (tid >= ROWS * NCG) return;
  int row = tid >> 6, j = tid & 63, b = row >> 10;
  Mn[tid] = M[tid] / colsum[b * NCG + j];
}

// ---- H[b,j,f] = sum_n Mn[b,n,j] * h[b,n,f] ----
__global__ void k_H(const float* __restrict__ Mn, const float* __restrict__ h,
                    float* __restrict__ H) {
  int tid = blockIdx.x * blockDim.x + threadIdx.x;
  if (tid >= BB * NCG * FF) return;
  int f = tid & 127, j = (tid >> 7) & 63, b = tid >> 13;
  const float* mn = Mn + (size_t)b * NN * NCG + j;
  const float* hh = h  + (size_t)b * NN * FF + f;
  float acc = 0.0f;
  for (int n = 0; n < NN; ++n) acc += mn[(size_t)n * NCG] * hh[(size_t)n * FF];
  H[tid] = acc;
}

// ---- cg_xyz[b,j,:] = sum_n Mn[b,n,j] * xyz[b,n,:] ----
__global__ void k_cgxyz(const float* __restrict__ xyz, const float* __restrict__ Mn,
                        float* __restrict__ cg) {
  int tid = blockIdx.x * blockDim.x + threadIdx.x;
  if (tid >= BB * NCG) return;
  int j = tid & 63, b = tid >> 6;
  const float* mn = Mn + (size_t)b * NN * NCG + j;
  const float* xp = xyz + (size_t)b * NN * 3;
  float a0 = 0, a1 = 0, a2 = 0;
  for (int n = 0; n < NN; ++n) {
    float w = mn[(size_t)n * NCG];
    a0 += w * xp[3 * n]; a1 += w * xp[3 * n + 1]; a2 += w * xp[3 * n + 2];
  }
  cg[tid * 3 + 0] = a0; cg[tid * 3 + 1] = a1; cg[tid * 3 + 2] = a2;
}

__global__ void k_cgadj(float* __restrict__ out) {
  int tid = blockIdx.x * blockDim.x + threadIdx.x;
  if (tid >= BB * NCG * NCG) return;
  int jj = tid & 63, ii = (tid >> 6) & 63;
  out[tid] = (ii == jj) ? 0.0f : 1.0f;
}

// ---- pairwise distances + stable argsort (indices written as float) ----
__global__ void k_knn(const float* __restrict__ cg, float* __restrict__ kn) {
  int tid = blockIdx.x * blockDim.x + threadIdx.x;
  if (tid >= BB * NCG) return;
  int i = tid & 63, b = tid >> 6;
  const float* base = cg + (size_t)b * NCG * 3;
  float xi = base[3 * i], yi = base[3 * i + 1], zi = base[3 * i + 2];
  float d[NCG];
  bool used[NCG];
  for (int j = 0; j < NCG; ++j) {
    float dx = xi - base[3 * j], dy = yi - base[3 * j + 1], dz = zi - base[3 * j + 2];
    d[j] = sqrtf(dx * dx + dy * dy + dz * dz + EPSV);
    used[j] = false;
  }
  for (int k = 0; k < NCG; ++k) {
    int best = -1; float bd = 3.0e38f;
    for (int j = 0; j < NCG; ++j)
      if (!used[j] && d[j] < bd) { bd = d[j]; best = j; }
    used[best] = true;
    kn[(size_t)tid * NCG + k] = (float)best;
  }
}

// -------------------------------------------------------------------------
extern "C" void kernel_launch(void* const* d_in, const int* in_sizes, int n_in,
                              void* d_out, int out_size, void* d_ws, size_t ws_size,
                              hipStream_t stream) {
  const int*   atoms = (const int*)  d_in[0];
  const float* xyz   = (const float*)d_in[1];
  const int*   bonds = (const int*)  d_in[2];
  const float* tau   = (const float*)d_in[3];
  const float* gu    = (const float*)d_in[4];
  const float* emb   = (const float*)d_in[5];
  const float* uW1   = (const float*)d_in[6];
  const float* ub1   = (const float*)d_in[7];
  const float* uW2   = (const float*)d_in[8];
  const float* ub2   = (const float*)d_in[9];
  const float* cW1   = (const float*)d_in[10];
  const float* cb1   = (const float*)d_in[11];
  const float* cW2   = (const float*)d_in[12];
  const float* cb2   = (const float*)d_in[13];
  const int nb = in_sizes[2] / 3;

  float* out = (float*)d_out;
  // output regions (flat float offsets, return order)
  const size_t OFF_M   = 0;
  const size_t OFF_MN  = OFF_M   + (size_t)ROWS * NCG;        // 1,048,576
  const size_t OFF_H   = OFF_MN  + (size_t)ROWS * NCG;        // 2,097,152
  const size_t OFF_HH  = OFF_H   + (size_t)ROWS * FF;         // 4,194,304
  const size_t OFF_ADJ = OFF_HH  + (size_t)BB * NCG * FF;     // 4,325,376
  const size_t OFF_CGX = OFF_ADJ + (size_t)BB * NN * NN;      // 21,102,592
  const size_t OFF_CGA = OFF_CGX + (size_t)BB * NCG * 3;      // 21,105,664
  const size_t OFF_KNN = OFF_CGA + (size_t)BB * NCG * NCG;    // 21,171,200

  // workspace carving
  char* ws = (char*)d_ws;
  float*     h32   = (float*)(ws + 0);                         //  8 MB
  float*     DH    = (float*)(ws + (8u << 20));                //  8 MB
  float*     CH    = (float*)(ws + (16u << 20));               //  8 MB
  float*     LOG   = (float*)(ws + (24u << 20));               //  4 MB
  _Float16*  h16   = (_Float16*)(ws + (28u << 20));            //  4 MB
  _Float16*  T16   = (_Float16*)(ws + (32u << 20));            //  4 MB
  _Float16*  W1p   = (_Float16*)(ws + (36u << 20));            //  96 KB
  _Float16*  W2p   = W1p + NCONV * FF * FF;
  _Float16*  cW1p  = W2p + NCONV * FF * FF;
  _Float16*  cW2p  = cW1p + FF * FF;
  float*     deg   = (float*)(cW2p + FF * NCG);
  float*     colsum = deg + BB * NN;

  // zero accumulators + dense adj output region
  hipMemsetAsync(deg, 0, BB * NN * sizeof(float), stream);
  hipMemsetAsync(colsum, 0, BB * NCG * sizeof(float), stream);
  hipMemsetAsync(out + OFF_ADJ, 0, (size_t)BB * NN * NN * sizeof(float), stream);

  // pre-pack all weights into WMMA-B lane layout (f16)
  for (int l = 0; l < NCONV; ++l) {
    k_pack_w<<<(FF * FF + 255) / 256, 256, 0, stream>>>(uW1 + (size_t)l * FF * FF,
                                                        W1p + (size_t)l * FF * FF, FF, FF);
    k_pack_w<<<(FF * FF + 255) / 256, 256, 0, stream>>>(uW2 + (size_t)l * FF * FF,
                                                        W2p + (size_t)l * FF * FF, FF, FF);
  }
  k_pack_w<<<(FF * FF + 255) / 256, 256, 0, stream>>>(cW1, cW1p, FF, FF);
  k_pack_w<<<(FF * NCG + 255) / 256, 256, 0, stream>>>(cW2, cW2p, FF, NCG);

  // embedding gather + dense adjacency/degree
  k_embed<<<(ROWS * FF) / 256, 256, 0, stream>>>(atoms, emb, h32, h16);
  k_adj<<<(nb + 255) / 256, 256, 0, stream>>>(bonds, nb, out + OFF_ADJ, deg);

  // graph-conv layers: WMMA GEMMs + bond-scatter aggregation
  const int tilesF = FF / 16;             // 8
  const int gemmBlocks = (ROWS / 16) * tilesF;  // 8192 waves
  for (int l = 0; l < NCONV; ++l) {
    hipMemsetAsync(DH, 0, (size_t)ROWS * FF * sizeof(float), stream);
    k_gemm_wmma<1><<<gemmBlocks, 32, 0, stream>>>(h16, W1p + (size_t)l * FF * FF,
                                                  ub1 + (size_t)l * FF,
                                                  nullptr, T16, tilesF, FF, FF);
    k_gemm_wmma<0><<<gemmBlocks, 32, 0, stream>>>(T16, W2p + (size_t)l * FF * FF,
                                                  ub2 + (size_t)l * FF,
                                                  CH, nullptr, tilesF, FF, FF);
    k_scatter<<<(nb * FF + 255) / 256, 256, 0, stream>>>(bonds, nb, CH, DH);
    k_update<<<(ROWS * FF) / 256, 256, 0, stream>>>(h32, h16, DH, deg);
  }

  // final h to output
  hipMemcpyAsync(out + OFF_H, h32, (size_t)ROWS * FF * sizeof(float),
                 hipMemcpyDeviceToDevice, stream);

  // assignment head: tanh(h@cgW1+b) @ cgW2 + b -> logits
  k_gemm_wmma<1><<<gemmBlocks, 32, 0, stream>>>(h16, cW1p, cb1, nullptr, T16,
                                                tilesF, FF, FF);
  k_gemm_wmma<0><<<(ROWS / 16) * (NCG / 16), 32, 0, stream>>>(T16, cW2p, cb2, LOG,
                                                              nullptr, NCG / 16, FF, NCG);

  // gumbel-softmax, column-normalize, pooling, distances, argsort
  k_softmax<<<ROWS / 256, 256, 0, stream>>>(LOG, gu, tau, out + OFF_M, colsum);
  k_mnorm<<<(ROWS * NCG) / 256, 256, 0, stream>>>(out + OFF_M, colsum, out + OFF_MN);
  k_H<<<(BB * NCG * FF) / 256, 256, 0, stream>>>(out + OFF_MN, h32, out + OFF_HH);
  k_cgxyz<<<(BB * NCG + 255) / 256, 256, 0, stream>>>(xyz, out + OFF_MN, out + OFF_CGX);
  k_cgadj<<<(BB * NCG * NCG) / 256, 256, 0, stream>>>(out + OFF_CGA);
  k_knn<<<(BB * NCG + 255) / 256, 256, 0, stream>>>(out + OFF_CGX, out + OFF_KNN);
  (void)n_in; (void)out_size; (void)ws_size;
}
